// MCI_GRU_Model_42322607735491
// MI455X (gfx1250) — compile-verified, matrix operand order checked by reading, not run on hardware
//
#include <hip/hip_runtime.h>
#include <hip/hip_bf16.h>

typedef __bf16 bf16;
typedef __attribute__((ext_vector_type(16))) __bf16 v16bf;
typedef __attribute__((ext_vector_type(8)))  float  v8f;

#define DEVINL __device__ __forceinline__

namespace {
constexpr int Bn  = 64;
constexpr int Tn  = 512;
constexpr int DINn = 64;
constexpr int Hn  = 256;
constexpr int NHn = 4;
constexpr int DKn = 64;
}

// ---------------------------------------------------------------------------
// WMMA helpers (CDNA5 16x16x32 bf16, f32 accumulate)
// A fragment: 16x32 (MxK), M = lane%16, K layout per ISA table:
//   VGPR v (pair e=2v,2v+1): K = (v>=4?16:0) + (lane>=16?8:0) + 2*(v%4) + half
// B^T fragment uses identical layout with N in place of M, reading BT[N,K].
// ---------------------------------------------------------------------------
DEVINL v16bf load_frag(const bf16* base, int ld) {
  const int lane = threadIdx.x & 31;
  const int row  = lane & 15;
  const int kb   = (lane & 16) ? 8 : 0;
  const bf16* rp = base + (size_t)row * ld + kb;
  v16bf f;
#pragma unroll
  for (int v = 0; v < 8; ++v) {
    const int k = ((v & 4) << 2) + ((v & 3) << 1);
    f[2 * v]     = rp[k];
    f[2 * v + 1] = rp[k + 1];
  }
  return f;
}

DEVINL v8f wmma_bf16(v16bf a, v16bf b, v8f c) {
  return __builtin_amdgcn_wmma_f32_16x16x32_bf16(
      /*neg_a=*/false, a, /*neg_b=*/false, b,
      /*c_mod=*/(short)0, c, /*reuse_a=*/false, /*reuse_b=*/false);
}

// C/D fragment: lanes 0-15 -> M = r, lanes 16-31 -> M = r+8 ; N = lane%16
DEVINL void store_tile(const v8f& c, float* Cf, bf16* Cb, const float* bias,
                       int ldc, int m0, int n0) {
  const int lane = threadIdx.x & 31;
  const int col  = lane & 15;
  const int rb   = (lane & 16) ? 8 : 0;
  const int n    = n0 + col;
  const float bv = bias ? bias[n] : 0.0f;
#pragma unroll
  for (int r = 0; r < 8; ++r) {
    const int m = m0 + rb + r;
    const float v = c[r] + bv;
    if (Cf) Cf[(size_t)m * ldc + n] = v;
    if (Cb) Cb[(size_t)m * ldc + n] = (bf16)v;
  }
}

// ---------------------------------------------------------------------------
// Generic WMMA GEMM:  C[M,N] = A[M,K] (bf16) * BT[N,K]^T (bf16) + bias
// 256 threads = 8 waves; each wave owns a 16x64 strip: one A fragment feeds
// 4 accumulators -> 4 back-to-back v_wmma per K block (XDL-friendly).
// Requires N % 64 == 0 (always 256 here).
// ---------------------------------------------------------------------------
__global__ void __launch_bounds__(256) wmma_gemm_kernel(
    const bf16* __restrict__ A, int lda,
    const bf16* __restrict__ BT, int ldb,
    const float* __restrict__ bias,
    float* __restrict__ Cf, bf16* __restrict__ Cb, int ldc,
    int M, int N, int K) {
  const int wave   = threadIdx.x >> 5;
  const int tilesN = N >> 6;  // 16x64 strips
  const int tile   = blockIdx.x * 8 + wave;
  if (tile >= (M >> 4) * tilesN) return;
  const int tm = tile / tilesN;
  const int tn = tile - tm * tilesN;
  const bf16* ap = A + (size_t)tm * 16 * lda;
  const bf16* bp = BT + (size_t)tn * 64 * ldb;
  v8f acc0 = {}, acc1 = {}, acc2 = {}, acc3 = {};
  for (int k = 0; k < K; k += 32) {
    __builtin_prefetch(ap + k + 64, 0, 1);
    __builtin_prefetch(bp + k + 64, 0, 1);
    v16bf a  = load_frag(ap + k, lda);
    v16bf b0 = load_frag(bp + k, ldb);
    v16bf b1 = load_frag(bp + (size_t)16 * ldb + k, ldb);
    v16bf b2 = load_frag(bp + (size_t)32 * ldb + k, ldb);
    v16bf b3 = load_frag(bp + (size_t)48 * ldb + k, ldb);
    acc0 = wmma_bf16(a, b0, acc0);
    acc1 = wmma_bf16(a, b1, acc1);
    acc2 = wmma_bf16(a, b2, acc2);
    acc3 = wmma_bf16(a, b3, acc3);
  }
  store_tile(acc0, Cf, Cb, bias, ldc, tm * 16, tn * 64);
  store_tile(acc1, Cf, Cb, bias, ldc, tm * 16, tn * 64 + 16);
  store_tile(acc2, Cf, Cb, bias, ldc, tm * 16, tn * 64 + 32);
  store_tile(acc3, Cf, Cb, bias, ldc, tm * 16, tn * 64 + 48);
}

// ---------------------------------------------------------------------------
// Persistent GRU scan. 4 blocks x 16 batch rows; h lives in LDS.
//   z = sigmoid(xz_t + h@Uz)            (bz folded into xz)
//   a = tanh(xa_t + h@Ua); w = softmax(a*va); att = w*h
//   ht = tanh(xh_t + att@Uh)            (bh folded into xh)
//   h  = (1-z)*h + z*ht
// A fragments (from h / attended in LDS) are loop-invariant across a wave's
// N tiles -> cached once per phase in registers (8 x v16bf = 64 VGPRs).
// ---------------------------------------------------------------------------
__global__ void __launch_bounds__(256) gru_scan_kernel(
    const float* __restrict__ xz, const float* __restrict__ xa,
    const float* __restrict__ xh,
    const bf16* __restrict__ UzT, const bf16* __restrict__ UaT,
    const bf16* __restrict__ UhT,
    const float* __restrict__ va,
    float* __restrict__ hout) {
  __shared__ float h_f[16][Hn];
  __shared__ bf16  h_b[16][Hn];
  __shared__ float zpre[16][Hn];
  __shared__ float apre[16][Hn];
  __shared__ float zval[16][Hn];
  __shared__ bf16  att_b[16][Hn];

  const int wave = threadIdx.x >> 5;
  const int lane = threadIdx.x & 31;
  const int b0   = blockIdx.x * 16;

  for (int i = threadIdx.x; i < 16 * Hn; i += 256) {
    (&h_f[0][0])[i] = 0.0f;
    (&h_b[0][0])[i] = (bf16)0.0f;
  }
  __syncthreads();

#pragma unroll 1
  for (int t = 0; t < Tn; ++t) {
    // ---- matmul phase 1: h@Uz -> zpre, h@Ua -> apre (32 tile jobs) ----
    {
      v16bf afr[8];
#pragma unroll
      for (int kk = 0; kk < 8; ++kk)
        afr[kk] = load_frag(&h_b[0][0] + kk * 32, Hn);
#pragma unroll 1
      for (int j = wave * 4; j < wave * 4 + 4; ++j) {
        const int mat = j >> 4;
        const int nt  = j & 15;
        const bf16*  UT  = mat ? UaT : UzT;
        const float* xin = mat ? xa : xz;
        float*       dst = mat ? &apre[0][0] : &zpre[0][0];
        v8f acc = {};
#pragma unroll
        for (int kk = 0; kk < 8; ++kk) {
          v16bf bb = load_frag(UT + (size_t)nt * 16 * Hn + kk * 32, Hn);
          acc = wmma_bf16(afr[kk], bb, acc);
        }
        const int col = lane & 15;
        const int rb  = (lane & 16) ? 8 : 0;
        const int n   = nt * 16 + col;
#pragma unroll
        for (int r = 0; r < 8; ++r) {
          const int m = rb + r;
          dst[m * Hn + n] =
              acc[r] + xin[((size_t)(b0 + m) * Tn + t) * Hn + n];
        }
      }
    }
    __syncthreads();

    // ---- elementwise: sigmoid(z), softmax(tanh(a)*va), att = w*h ----
#pragma unroll 1
    for (int mi = 0; mi < 2; ++mi) {
      const int m = wave * 2 + mi;
      float sv[8];
      float mx = -3.402823466e38f;
#pragma unroll
      for (int jj = 0; jj < 8; ++jj) {
        const int n  = lane + jj * 32;
        const float s = tanhf(apre[m][n]) * va[n];
        sv[jj] = s;
        mx = fmaxf(mx, s);
      }
      for (int o = 16; o; o >>= 1) mx = fmaxf(mx, __shfl_xor(mx, o));
      float sum = 0.0f;
#pragma unroll
      for (int jj = 0; jj < 8; ++jj) {
        sv[jj] = __expf(sv[jj] - mx);
        sum += sv[jj];
      }
      for (int o = 16; o; o >>= 1) sum += __shfl_xor(sum, o);
      const float inv = 1.0f / sum;
#pragma unroll
      for (int jj = 0; jj < 8; ++jj) {
        const int n = lane + jj * 32;
        att_b[m][n] = (bf16)(sv[jj] * inv * h_f[m][n]);
        zval[m][n]  = 1.0f / (1.0f + __expf(-zpre[m][n]));
      }
    }
    __syncthreads();

    // ---- matmul phase 2: att@Uh -> h update (16 tile jobs) ----
    {
      v16bf afr[8];
#pragma unroll
      for (int kk = 0; kk < 8; ++kk)
        afr[kk] = load_frag(&att_b[0][0] + kk * 32, Hn);
#pragma unroll 1
      for (int j2 = 0; j2 < 2; ++j2) {
        const int nt = wave * 2 + j2;
        v8f acc = {};
#pragma unroll
        for (int kk = 0; kk < 8; ++kk) {
          v16bf bb = load_frag(UhT + (size_t)nt * 16 * Hn + kk * 32, Hn);
          acc = wmma_bf16(afr[kk], bb, acc);
        }
        const int col = lane & 15;
        const int rb  = (lane & 16) ? 8 : 0;
        const int n   = nt * 16 + col;
#pragma unroll
        for (int r = 0; r < 8; ++r) {
          const int m  = rb + r;
          const float ht =
              tanhf(acc[r] + xh[((size_t)(b0 + m) * Tn + t) * Hn + n]);
          const float z  = zval[m][n];
          const float hn = (1.0f - z) * h_f[m][n] + z * ht;
          hout[((size_t)(b0 + m) * Tn + t) * Hn + n] = hn;
          h_f[m][n] = hn;
          h_b[m][n] = (bf16)hn;
        }
      }
    }
    __syncthreads();
  }
}

// ---------------------------------------------------------------------------
// LayerNorm over rows of [rows, 256]; one wave per row.
// ---------------------------------------------------------------------------
__global__ void __launch_bounds__(256) layernorm_kernel(
    const float* __restrict__ X, const float* __restrict__ g,
    const float* __restrict__ bta, float* __restrict__ Yf,
    bf16* __restrict__ Yb, int rows) {
  const int row = blockIdx.x * 8 + (threadIdx.x >> 5);
  if (row >= rows) return;
  const int lane = threadIdx.x & 31;
  const float* x = X + (size_t)row * Hn;
  float sum = 0.0f;
  for (int j = lane; j < Hn; j += 32) sum += x[j];
  for (int o = 16; o; o >>= 1) sum += __shfl_xor(sum, o);
  const float mu = sum * (1.0f / Hn);
  float vs = 0.0f;
  for (int j = lane; j < Hn; j += 32) {
    const float d = x[j] - mu;
    vs += d * d;
  }
  for (int o = 16; o; o >>= 1) vs += __shfl_xor(vs, o);
  const float inv = rsqrtf(vs * (1.0f / Hn) + 1e-5f);
  for (int j = lane; j < Hn; j += 32) {
    const float y = (x[j] - mu) * inv * g[j] + bta[j];
    if (Yf) Yf[(size_t)row * Hn + j] = y;
    if (Yb) Yb[(size_t)row * Hn + j] = (bf16)y;
  }
}

// ---------------------------------------------------------------------------
// Cross attention: per (qtile, head, batch) block.
// scores -> softmax -> aw (global, f32) ; only last qtile computes aw@V row 511.
// ---------------------------------------------------------------------------
__global__ void __launch_bounds__(256) attn_kernel(
    const bf16* __restrict__ Qb, const bf16* __restrict__ Kb,
    const bf16* __restrict__ Vt,   // [B, NH, DK, T] bf16
    float* __restrict__ aw,        // [B, NH, T, T]
    float* __restrict__ outlast) { // [B, H]
  __shared__ float s[16][Tn];
  __shared__ bf16  awb[16][Tn];
  const int qt   = blockIdx.x;  // 0..31
  const int h    = blockIdx.y;  // 0..3
  const int b    = blockIdx.z;  // 0..63
  const int wave = threadIdx.x >> 5;
  const int lane = threadIdx.x & 31;

  const bf16* Qbase = Qb + ((size_t)b * Tn + qt * 16) * Hn + h * DKn;
  const bf16* Kbase = Kb + (size_t)b * Tn * Hn + h * DKn;

  // phase 1: 16x512 score strip, scaled by 1/sqrt(DK).
  // Q fragments are loop invariant -> hoisted.
  {
    const v16bf a0 = load_frag(Qbase, Hn);
    const v16bf a1 = load_frag(Qbase + 32, Hn);
#pragma unroll 1
    for (int kt = wave; kt < Tn / 16; kt += 8) {
      const bf16* kb = Kbase + (size_t)kt * 16 * Hn;
      v8f acc = {};
      acc = wmma_bf16(a0, load_frag(kb, Hn), acc);
      acc = wmma_bf16(a1, load_frag(kb + 32, Hn), acc);
      const int col = lane & 15;
      const int rb  = (lane & 16) ? 8 : 0;
#pragma unroll
      for (int r = 0; r < 8; ++r) s[rb + r][kt * 16 + col] = acc[r] * 0.125f;
    }
  }
  __syncthreads();

  // phase 2: row softmax (one wave per 2 rows), write aw + bf16 copy in LDS
#pragma unroll 1
  for (int mi = 0; mi < 2; ++mi) {
    const int m = wave * 2 + mi;
    float mx = -3.402823466e38f;
    for (int j = lane; j < Tn; j += 32) mx = fmaxf(mx, s[m][j]);
    for (int o = 16; o; o >>= 1) mx = fmaxf(mx, __shfl_xor(mx, o));
    float sum = 0.0f;
    for (int j = lane; j < Tn; j += 32) {
      const float e = __expf(s[m][j] - mx);
      s[m][j] = e;
      sum += e;
    }
    for (int o = 16; o; o >>= 1) sum += __shfl_xor(sum, o);
    const float inv = 1.0f / sum;
    float* awrow = aw + (((size_t)b * NHn + h) * Tn + qt * 16 + m) * Tn;
    for (int j = lane; j < Tn; j += 32) {
      const float w = s[m][j] * inv;
      awrow[j]  = w;
      awb[m][j] = (bf16)w;
    }
  }
  __syncthreads();

  // phase 3: only tile containing q = T-1 computes aw@V (row 15)
  if (qt == Tn / 16 - 1 && wave < 4) {
    v8f acc = {};
    const bf16* vb = Vt + (((size_t)b * NHn + h) * DKn + wave * 16) * Tn;
#pragma unroll 1
    for (int k = 0; k < Tn; k += 32) {
      v16bf a  = load_frag(&awb[0][0] + k, Tn);
      v16bf bb = load_frag(vb + k, Tn);
      acc = wmma_bf16(a, bb, acc);
    }
    if (lane >= 16) { // rows 8..15: r=7 -> m=15 -> q = T-1
      outlast[(size_t)b * Hn + h * DKn + wave * 16 + (lane & 15)] = acc[7];
    }
  }
}

// ---------------------------------------------------------------------------
// Small utility kernels
// ---------------------------------------------------------------------------
__global__ void transpose_w_kernel(const float* __restrict__ W,
                                   bf16* __restrict__ WT, int K, int N) {
  const int i = blockIdx.x * 256 + threadIdx.x;
  if (i >= K * N) return;
  const int n = i / K;
  const int k = i - n * K;
  WT[i] = (bf16)W[(size_t)k * N + n];
}

__global__ void cvt_bf16_kernel(const float* __restrict__ in,
                                bf16* __restrict__ out, int n) {
  const int i = blockIdx.x * 256 + threadIdx.x;
  if (i < n) out[i] = (bf16)in[i];
}

__global__ void vtrans_kernel(const float* __restrict__ V,  // [B,T,H]
                              bf16* __restrict__ Vt) {       // [B,NH,DK,T]
  const size_t i = (size_t)blockIdx.x * 256 + threadIdx.x;
  if (i >= (size_t)Bn * NHn * DKn * Tn) return;
  const int t = (int)(i % Tn);
  size_t r = i / Tn;
  const int d = (int)(r % DKn); r /= DKn;
  const int h = (int)(r % NHn);
  const int b = (int)(r / NHn);
  Vt[i] = (bf16)V[((size_t)b * Tn + t) * Hn + h * DKn + d];
}

__global__ void add_last_kernel(const float* __restrict__ woOut,
                                const float* __restrict__ g2nf,
                                float* __restrict__ feats) {
  const int i = blockIdx.x * 256 + threadIdx.x;
  if (i >= Bn * Hn) return;
  const int b = i >> 8;
  const int n = i & 255;
  feats[i] = woOut[i] + g2nf[((size_t)b * Tn + (Tn - 1)) * Hn + n];
}

__global__ void fc_kernel(const float* __restrict__ X, const float* __restrict__ W,
                          const float* __restrict__ bias, float* __restrict__ Y,
                          int M, int K, int N, int relu) {
  const int i = blockIdx.x * blockDim.x + threadIdx.x;
  if (i >= M * N) return;
  const int m = i / N;
  const int n = i - m * N;
  float s = bias[n];
  for (int k = 0; k < K; ++k) s += X[(size_t)m * K + k] * W[(size_t)k * N + n];
  Y[i] = relu ? fmaxf(s, 0.0f) : s;
}

// ---------------------------------------------------------------------------
// Host orchestration
// ---------------------------------------------------------------------------
extern "C" void kernel_launch(void* const* d_in, const int* in_sizes, int n_in,
                              void* d_out, int out_size, void* d_ws, size_t ws_size,
                              hipStream_t stream) {
  (void)in_sizes; (void)n_in; (void)out_size; (void)ws_size;
  const float* F[41];
  for (int i = 0; i < 41; ++i) F[i] = (const float*)d_in[i];

  // input indices (setup_inputs dict insertion order)
  const float* x    = F[0];
  const float *g1Wz = F[1], *g1Uz = F[2], *g1bz = F[3], *g1Wa = F[4], *g1Ua = F[5],
              *g1va = F[6], *g1Wh = F[7], *g1Uh = F[8], *g1bh = F[9];
  const float *g2Wz = F[10], *g2Uz = F[11], *g2bz = F[12], *g2Wa = F[13],
              *g2Ua = F[14], *g2va = F[15], *g2Wh = F[16], *g2Uh = F[17], *g2bh = F[18];
  const float *Wq = F[19], *bq = F[20], *Wk = F[21], *bk = F[22],
              *Wv = F[23], *bv = F[24], *Wo = F[25], *bo = F[26];
  const float *ln1g = F[27], *ln1b = F[28], *ln2g = F[29], *ln2b = F[30];
  const float *Wt = F[31], *bt = F[32], *Wc = F[33], *bc = F[34];
  const float *W1 = F[35], *b1 = F[36], *W2 = F[37], *b2 = F[38],
              *W3 = F[39], *b3 = F[40];

  const size_t BT  = (size_t)Bn * Tn;
  const size_t BTH = BT * Hn;

  char* p = (char*)d_ws;
  auto alloc = [&](size_t bytes) {
    char* r = p;
    p += (bytes + 255) & ~(size_t)255;
    return (void*)r;
  };

  bf16* x_bf   = (bf16*)alloc(BT * DINn * sizeof(bf16));
  bf16* g1WzT  = (bf16*)alloc((size_t)DINn * Hn * sizeof(bf16));
  bf16* g1WaT  = (bf16*)alloc((size_t)DINn * Hn * sizeof(bf16));
  bf16* g1WhT  = (bf16*)alloc((size_t)DINn * Hn * sizeof(bf16));
  bf16* g1UzT  = (bf16*)alloc((size_t)Hn * Hn * sizeof(bf16));
  bf16* g1UaT  = (bf16*)alloc((size_t)Hn * Hn * sizeof(bf16));
  bf16* g1UhT  = (bf16*)alloc((size_t)Hn * Hn * sizeof(bf16));
  bf16* g2WzT  = (bf16*)alloc((size_t)Hn * Hn * sizeof(bf16));
  bf16* g2WaT  = (bf16*)alloc((size_t)Hn * Hn * sizeof(bf16));
  bf16* g2WhT  = (bf16*)alloc((size_t)Hn * Hn * sizeof(bf16));
  bf16* g2UzT  = (bf16*)alloc((size_t)Hn * Hn * sizeof(bf16));
  bf16* g2UaT  = (bf16*)alloc((size_t)Hn * Hn * sizeof(bf16));
  bf16* g2UhT  = (bf16*)alloc((size_t)Hn * Hn * sizeof(bf16));
  bf16* WqT    = (bf16*)alloc((size_t)Hn * Hn * sizeof(bf16));
  bf16* WkT    = (bf16*)alloc((size_t)Hn * Hn * sizeof(bf16));
  bf16* WvT    = (bf16*)alloc((size_t)Hn * Hn * sizeof(bf16));
  bf16* WoT    = (bf16*)alloc((size_t)Hn * Hn * sizeof(bf16));
  bf16* WtT    = (bf16*)alloc((size_t)Hn * Hn * sizeof(bf16));
  bf16* WcT    = (bf16*)alloc((size_t)Hn * Hn * sizeof(bf16));
  float* bufXz = (float*)alloc(BTH * sizeof(float));
  float* bufXa = (float*)alloc(BTH * sizeof(float));
  float* bufXh = (float*)alloc(BTH * sizeof(float));
  float* bufG  = (float*)alloc(BTH * sizeof(float));
  bf16* ln1_bf = (bf16*)alloc(BTH * sizeof(bf16));
  bf16* ln2_bf = (bf16*)alloc(BTH * sizeof(bf16));
  float* outlast    = (float*)alloc((size_t)Bn * Hn * sizeof(float));
  bf16*  outlast_bf = (bf16*)alloc((size_t)Bn * Hn * sizeof(bf16));
  float* woOut = (float*)alloc((size_t)Bn * Hn * sizeof(float));
  float* feats = (float*)alloc((size_t)Bn * Hn * sizeof(float));
  float* h1    = (float*)alloc((size_t)Bn * 128 * sizeof(float));
  float* h2    = (float*)alloc((size_t)Bn * 64 * sizeof(float));

  // reuse views (lifetimes verified against pipeline order)
  float* g2n_f       = bufXz;                 // after GRU2 scan
  bf16*  temporal_bf = (bf16*)bufXa;          // after GRU2 scan
  bf16*  Qbf         = (bf16*)bufXa + BTH;    // second half of bufXa
  bf16*  cross_bf    = (bf16*)bufXh;
  bf16*  Kbf         = (bf16*)bufXh + BTH;
  float* Vf          = bufG;                  // after ln2
  bf16*  Vtb         = ln1_bf;                // after GRU2 projections

  float* out_part = (float*)d_out;        // [64,1]
  float* aw_part  = (float*)d_out + 64;   // [64,4,512,512]

  auto gemm = [&](const bf16* A, int lda, const bf16* BTw, int ldb,
                  const float* bias, float* Cf, bf16* Cb, int ldc,
                  int M, int N, int K) {
    const int tiles  = (M / 16) * (N / 64);
    const int blocks = (tiles + 7) / 8;
    wmma_gemm_kernel<<<blocks, 256, 0, stream>>>(A, lda, BTw, ldb, bias, Cf, Cb,
                                                 ldc, M, N, K);
  };
  auto trw = [&](const float* W, bf16* WT, int K, int N) {
    transpose_w_kernel<<<(K * N + 255) / 256, 256, 0, stream>>>(W, WT, K, N);
  };

  // --- stage 0: conversions ---
  cvt_bf16_kernel<<<(int)((BT * DINn + 255) / 256), 256, 0, stream>>>(
      x, x_bf, (int)(BT * DINn));
  trw(g1Wz, g1WzT, DINn, Hn); trw(g1Wa, g1WaT, DINn, Hn); trw(g1Wh, g1WhT, DINn, Hn);
  trw(g1Uz, g1UzT, Hn, Hn);   trw(g1Ua, g1UaT, Hn, Hn);   trw(g1Uh, g1UhT, Hn, Hn);
  trw(g2Wz, g2WzT, Hn, Hn);   trw(g2Wa, g2WaT, Hn, Hn);   trw(g2Wh, g2WhT, Hn, Hn);
  trw(g2Uz, g2UzT, Hn, Hn);   trw(g2Ua, g2UaT, Hn, Hn);   trw(g2Uh, g2UhT, Hn, Hn);
  trw(Wq, WqT, Hn, Hn); trw(Wk, WkT, Hn, Hn); trw(Wv, WvT, Hn, Hn); trw(Wo, WoT, Hn, Hn);
  trw(Wt, WtT, Hn, Hn); trw(Wc, WcT, Hn, Hn);

  // --- stage 1: GRU1 input projections (bz/bh folded in) ---
  gemm(x_bf, DINn, g1WzT, DINn, g1bz,   bufXz, nullptr, Hn, (int)BT, Hn, DINn);
  gemm(x_bf, DINn, g1WaT, DINn, nullptr, bufXa, nullptr, Hn, (int)BT, Hn, DINn);
  gemm(x_bf, DINn, g1WhT, DINn, g1bh,   bufXh, nullptr, Hn, (int)BT, Hn, DINn);

  // --- stage 2: GRU1 scan + LN1 ---
  gru_scan_kernel<<<Bn / 16, 256, 0, stream>>>(bufXz, bufXa, bufXh,
                                               g1UzT, g1UaT, g1UhT, g1va, bufG);
  layernorm_kernel<<<(int)(BT / 8), 256, 0, stream>>>(bufG, ln1g, ln1b,
                                                      nullptr, ln1_bf, (int)BT);

  // --- stage 3: GRU2 projections + scan + LN2 ---
  gemm(ln1_bf, Hn, g2WzT, Hn, g2bz,   bufXz, nullptr, Hn, (int)BT, Hn, Hn);
  gemm(ln1_bf, Hn, g2WaT, Hn, nullptr, bufXa, nullptr, Hn, (int)BT, Hn, Hn);
  gemm(ln1_bf, Hn, g2WhT, Hn, g2bh,   bufXh, nullptr, Hn, (int)BT, Hn, Hn);
  gru_scan_kernel<<<Bn / 16, 256, 0, stream>>>(bufXz, bufXa, bufXh,
                                               g2UzT, g2UaT, g2UhT, g2va, bufG);
  layernorm_kernel<<<(int)(BT / 8), 256, 0, stream>>>(bufG, ln2g, ln2b,
                                                      g2n_f, ln2_bf, (int)BT);

  // --- stage 4: temporal / cross / Q / K / V projections ---
  gemm(ln2_bf, Hn, WtT, Hn, bt, nullptr, temporal_bf, Hn, (int)BT, Hn, Hn);
  gemm(ln2_bf, Hn, WcT, Hn, bc, nullptr, cross_bf,    Hn, (int)BT, Hn, Hn);
  gemm(temporal_bf, Hn, WqT, Hn, bq, nullptr, Qbf, Hn, (int)BT, Hn, Hn);
  gemm(cross_bf,    Hn, WkT, Hn, bk, nullptr, Kbf, Hn, (int)BT, Hn, Hn);
  gemm(cross_bf,    Hn, WvT, Hn, bv, Vf, nullptr, Hn, (int)BT, Hn, Hn);
  vtrans_kernel<<<(int)(((size_t)Bn * NHn * DKn * Tn + 255) / 256), 256, 0,
                  stream>>>(Vf, Vtb);

  // --- stage 5: attention (writes aw to d_out, outlast for q = T-1) ---
  attn_kernel<<<dim3(Tn / 16, NHn, Bn), 256, 0, stream>>>(Qbf, Kbf, Vtb,
                                                          aw_part, outlast);

  // --- stage 6: Wo projection on last timestep + residual + FC head ---
  cvt_bf16_kernel<<<(Bn * Hn + 255) / 256, 256, 0, stream>>>(outlast, outlast_bf,
                                                             Bn * Hn);
  gemm(outlast_bf, Hn, WoT, Hn, bo, woOut, nullptr, Hn, Bn, Hn, Hn);
  add_last_kernel<<<(Bn * Hn + 255) / 256, 256, 0, stream>>>(woOut, g2n_f, feats);
  fc_kernel<<<(Bn * 128 + 255) / 256, 256, 0, stream>>>(feats, W1, b1, h1,
                                                        Bn, Hn, 128, 1);
  fc_kernel<<<(Bn * 64 + 255) / 256, 256, 0, stream>>>(h1, W2, b2, h2,
                                                       Bn, 128, 64, 1);
  fc_kernel<<<(Bn + 255) / 256, 256, 0, stream>>>(h2, W3, b3, out_part,
                                                  Bn, 64, 1, 0);
}
